// Transformer_24970939859612
// MI455X (gfx1250) — matrix operand order repaired and checked by reading, not compiled, not run on hardware
//
#include <hip/hip_runtime.h>
#include <hip/hip_bf16.h>

typedef __attribute__((ext_vector_type(16))) __bf16 v16bf;
typedef __attribute__((ext_vector_type(8)))  __bf16 v8bf;
typedef __attribute__((ext_vector_type(8)))  float  v8f;
typedef __attribute__((ext_vector_type(4)))  __bf16 v4bf;
typedef __attribute__((ext_vector_type(4)))  int    v4i_;

// pointer-to-int4 in specific address spaces (AS1 = global, AS3 = LDS)
typedef __attribute__((address_space(1))) v4i_* as1_v4i;
typedef __attribute__((address_space(3))) v4i_* as3_v4i;

#define UNITS   128
#define HEADS   8
#define D_IN    1024
#define BATCH   4
#define SEQ     2048
#define M_TOTAL (BATCH * SEQ)   // 8192

// ---------------------------------------------------------------------------
// CDNA5 async global->LDS copy (ASYNCcnt) with synchronous fallback
// ---------------------------------------------------------------------------
#if defined(__HIP_DEVICE_COMPILE__) &&                                         \
    __has_builtin(__builtin_amdgcn_global_load_async_to_lds_b128) &&           \
    __has_builtin(__builtin_amdgcn_s_wait_asynccnt)
#define USE_ASYNC_LDS 1
#else
#define USE_ASYNC_LDS 0
#endif

static __device__ __forceinline__ void cp16(void* ldsDst, const void* gSrc) {
#if USE_ASYNC_LDS
    __builtin_amdgcn_global_load_async_to_lds_b128(
        (as1_v4i)(gSrc), (as3_v4i)(ldsDst), 0, 0);
#else
    *(uint4*)ldsDst = *(const uint4*)gSrc;
#endif
}
static __device__ __forceinline__ void cp_wait() {
#if USE_ASYNC_LDS
    __builtin_amdgcn_s_wait_asynccnt(0);
#endif
}

// ---------------------------------------------------------------------------
// helpers
// ---------------------------------------------------------------------------
static __device__ __forceinline__ v8f vzero8() {
    v8f z;
#pragma unroll
    for (int i = 0; i < 8; ++i) z[i] = 0.0f;
    return z;
}

static __device__ __forceinline__ v16bf load_frag2(const __bf16* p0, const __bf16* p1) {
    union { uint4 u[2]; v16bf v; } r;
    r.u[0] = *(const uint4*)p0;
    r.u[1] = *(const uint4*)p1;
    return r.v;
}

// ---------------------------------------------------------------------------
// fp32 -> bf16 conversion (row major, bandwidth bound)
// ---------------------------------------------------------------------------
__global__ void cvt_f32_to_bf16(const float* __restrict__ in,
                                __bf16* __restrict__ out, int n4) {
    int i = blockIdx.x * blockDim.x + threadIdx.x;
    if (i < n4) {
        float4 f = ((const float4*)in)[i];
        v4bf o;
        o[0] = (__bf16)f.x; o[1] = (__bf16)f.y;
        o[2] = (__bf16)f.z; o[3] = (__bf16)f.w;
        ((v4bf*)out)[i] = o;
    }
}

// ---------------------------------------------------------------------------
// fp32 [z][K][N] -> bf16 [z][N][K] tiled transpose (weights, one-time)
// ---------------------------------------------------------------------------
__global__ __launch_bounds__(256) void transpose_cvt_bf16(
    const float* __restrict__ in, __bf16* __restrict__ out, int K, int N) {
    __shared__ float tile[32][33];
    const long inZ  = (long)blockIdx.z * K * N;
    const long outZ = (long)blockIdx.z * N * K;
    const int k0 = blockIdx.x * 32, n0 = blockIdx.y * 32;
    const int tx = threadIdx.x & 31, ty = threadIdx.x >> 5;
#pragma unroll
    for (int r = 0; r < 32; r += 8)
        tile[ty + r][tx] = in[inZ + (long)(k0 + ty + r) * N + n0 + tx];
    __syncthreads();
#pragma unroll
    for (int r = 0; r < 32; r += 8)
        out[outZ + (long)(n0 + ty + r) * K + k0 + tx] = (__bf16)tile[tx][ty + r];
}

// ---------------------------------------------------------------------------
// Tiled GEMM:  C[M x 128] = A[M x K](bf16) * W^T  (W stored [128][K] bf16) + bias
// grid.y indexes 24 (which,head) triples. BM=128,BN=128,BK=32, 8 waves (4x2).
// Double-buffered LDS, async global->LDS staging.
// For blockIdx.y >= 16 (V projection) the output is stored TRANSPOSED into
// vT[h][u][m] with packed 16B stores, so attention can async-copy V tiles.
// ---------------------------------------------------------------------------
template <bool OUT_BF16>
__global__ __launch_bounds__(256) void gemm_bias_n128(
    const __bf16* __restrict__ A, const __bf16* __restrict__ Wbase,
    const float* __restrict__ biasBase, void* __restrict__ Obase,
    __bf16* __restrict__ vTbase,
    int M, int K, long wStride, long bStride, long oStride) {
    const int tid  = threadIdx.x;
    const int lane = tid & 31;
    const int wid  = tid >> 5;
    const int mw   = wid >> 1;
    const int nw   = wid & 1;
    const int half = lane >> 4;
    const int l16  = lane & 15;
    const int m0   = blockIdx.x * 128;

    const __bf16* W    = Wbase + (long)blockIdx.y * wStride;   // [128][K]
    const float*  bias = biasBase + (long)blockIdx.y * bStride;

    __shared__ __bf16 ldsA[2][128][40];   // [m][k]
    __shared__ __bf16 ldsB[2][128][40];   // [n][k]

    v8f acc[2][4];
#pragma unroll
    for (int i = 0; i < 2; ++i)
#pragma unroll
        for (int j = 0; j < 4; ++j) acc[i][j] = vzero8();

    auto load_tiles = [&](int buf, int k0) {
#pragma unroll
        for (int j = 0; j < 2; ++j) {
            int i = tid + j * 256;           // A: 128 rows x 4 chunks
            int row = i >> 2, cg = i & 3;
            cp16(&ldsA[buf][row][cg * 8], A + (long)(m0 + row) * K + k0 + cg * 8);
        }
#pragma unroll
        for (int j = 0; j < 2; ++j) {
            int i = tid + j * 256;           // B: 128 n-rows x 4 chunks
            int n = i >> 2, cg = i & 3;
            cp16(&ldsB[buf][n][cg * 8], W + (long)n * K + k0 + cg * 8);
        }
    };

    load_tiles(0, 0);
    cp_wait();
    __syncthreads();

    int p = 0;
    for (int k0 = 0; k0 < K; k0 += 32) {
        if (k0 + 32 < K) load_tiles(p ^ 1, k0 + 32);

        v16bf af[2], bfr[4];
#pragma unroll
        for (int mf = 0; mf < 2; ++mf) {
            const __bf16* q = &ldsA[p][mw * 32 + mf * 16 + l16][half * 8];
            af[mf] = load_frag2(q, q + 16);
        }
#pragma unroll
        for (int nf = 0; nf < 4; ++nf) {
            const __bf16* q = &ldsB[p][nw * 64 + nf * 16 + l16][half * 8];
            bfr[nf] = load_frag2(q, q + 16);
        }
#pragma unroll
        for (int mf = 0; mf < 2; ++mf)
#pragma unroll
            for (int nf = 0; nf < 4; ++nf)
                acc[mf][nf] = __builtin_amdgcn_wmma_f32_16x16x32_bf16(
                    false, af[mf], false, bfr[nf], (short)0, acc[mf][nf], false, false);

        cp_wait();
        __syncthreads();
        p ^= 1;
    }

    // epilogue: C/D frag layout row = r + 8*half, col = l16
#pragma unroll
    for (int mf = 0; mf < 2; ++mf) {
#pragma unroll
        for (int nf = 0; nf < 4; ++nf) {
            int n = nw * 64 + nf * 16 + l16;
            float bv = bias[n];
            int row0 = m0 + mw * 32 + mf * 16 + half * 8;
            if constexpr (OUT_BF16) {
                if (blockIdx.y >= 16) {
                    // V plane: store transposed, 8 consecutive m -> 16B packed
                    __bf16* O = vTbase + (long)(blockIdx.y - 16) * 128 * M_TOTAL;
                    union { v8bf v; uint4 u; } pk;
#pragma unroll
                    for (int r = 0; r < 8; ++r) pk.v[r] = (__bf16)(acc[mf][nf][r] + bv);
                    *(uint4*)&O[(long)n * M_TOTAL + row0] = pk.u;
                } else {
                    __bf16* O = (__bf16*)Obase + (long)blockIdx.y * oStride;
#pragma unroll
                    for (int r = 0; r < 8; ++r)
                        O[(long)(row0 + r) * 128 + n] = (__bf16)(acc[mf][nf][r] + bv);
                }
            } else {
                float* O = (float*)Obase;
#pragma unroll
                for (int r = 0; r < 8; ++r)
                    O[(long)(row0 + r) * 128 + n] = acc[mf][nf][r] + bv;
            }
        }
    }
}

// ---------------------------------------------------------------------------
// Flash attention per (b,h). Q,K planes row-major [h][m][128]; V pre-transposed
// vT[h][u][m]. Key blocks of 64, double-buffered async tiles, online softmax
// with half-wave shfl reductions; P staged through per-wave LDS (same-wave DS
// ordering, no barrier). Output -> flat[B*S][H*128] bf16.
// ---------------------------------------------------------------------------
__global__ __launch_bounds__(256) void attn_kernel(
    const __bf16* __restrict__ qkPlanes,   // [2][H][M_TOTAL][128]
    const __bf16* __restrict__ vT,         // [H][128][M_TOTAL]
    __bf16* __restrict__ flat) {           // [M_TOTAL][H*128]
    const int tid  = threadIdx.x;
    const int lane = tid & 31;
    const int wid  = tid >> 5;
    const int half = lane >> 4;
    const int l16  = lane & 15;

    const int bh = blockIdx.y;
    const int b = bh >> 3, h = bh & 7;
    const long planeSz = (long)HEADS * M_TOTAL * UNITS;
    const long headOff = ((long)h * M_TOTAL + (long)b * SEQ) * UNITS;
    const __bf16* Qp = qkPlanes + headOff;
    const __bf16* Kp = qkPlanes + planeSz + headOff;
    const __bf16* Vt = vT + (long)h * UNITS * M_TOTAL + (long)b * SEQ;  // [u][key]

    __shared__ __bf16 ldsK[2][64][136];   // [key][u]
    __shared__ __bf16 ldsV[2][128][72];   // [u][key]
    __shared__ __bf16 ldsP[8][16][72];    // per-wave P staging [row][key]

    const float SCALE = 0.08838834764831845f;  // 1/sqrt(128)
    const float L2E   = 1.4426950408889634f;

    // Q fragments (A layout), scale folded in once
    const int srow = blockIdx.x * 128 + wid * 16 + l16;
    v16bf aq[4];
#pragma unroll
    for (int c = 0; c < 4; ++c) {
        const __bf16* q = Qp + (long)srow * 128 + c * 32 + half * 8;
        aq[c] = load_frag2(q, q + 16);
#pragma unroll
        for (int e = 0; e < 16; ++e) aq[c][e] = (__bf16)((float)aq[c][e] * SCALE);
    }

    v8f o[8];
#pragma unroll
    for (int uf = 0; uf < 8; ++uf) o[uf] = vzero8();
    float rm[8], rs[8];
#pragma unroll
    for (int r = 0; r < 8; ++r) { rm[r] = -1e30f; rs[r] = 0.0f; }

    auto load_tiles = [&](int buf, int kb) {
#pragma unroll
        for (int j = 0; j < 4; ++j) {
            int i = tid + j * 256;            // K: 64 rows x 16 chunks
            int row = i >> 4, cg = i & 15;
            cp16(&ldsK[buf][row][cg * 8], Kp + (long)(kb + row) * 128 + cg * 8);
        }
#pragma unroll
        for (int j = 0; j < 4; ++j) {
            int i = tid + j * 256;            // V: 128 u-rows x 8 chunks
            int u = i >> 3, cg = i & 7;
            cp16(&ldsV[buf][u][cg * 8], Vt + (long)u * M_TOTAL + kb + cg * 8);
        }
    };

    load_tiles(0, 0);
    cp_wait();
    __syncthreads();

    int p = 0;
    for (int kb = 0; kb < SEQ; kb += 64) {
        if (kb + 64 < SEQ) load_tiles(p ^ 1, kb + 64);

        // S = (Q*scale) K^T : 4 key-frags, U=128 in 4 slabs of 32
        v8f s[4];
#pragma unroll
        for (int nf = 0; nf < 4; ++nf) {
            s[nf] = vzero8();
#pragma unroll
            for (int c = 0; c < 4; ++c) {
                const __bf16* q = &ldsK[p][nf * 16 + l16][c * 32 + half * 8];
                v16bf bk = load_frag2(q, q + 16);
                s[nf] = __builtin_amdgcn_wmma_f32_16x16x32_bf16(
                    false, aq[c], false, bk, (short)0, s[nf], false, false);
            }
        }

        // online softmax
        float mnew[8], corr[8], psum[8];
#pragma unroll
        for (int r = 0; r < 8; ++r) {
            float cm = fmaxf(fmaxf(s[0][r], s[1][r]), fmaxf(s[2][r], s[3][r]));
            cm = fmaxf(cm, __shfl_xor(cm, 1));
            cm = fmaxf(cm, __shfl_xor(cm, 2));
            cm = fmaxf(cm, __shfl_xor(cm, 4));
            cm = fmaxf(cm, __shfl_xor(cm, 8));
            float mn = fmaxf(rm[r], cm);
            corr[r] = exp2f((rm[r] - mn) * L2E);
            rm[r] = mn; mnew[r] = mn; psum[r] = 0.0f;
        }
#pragma unroll
        for (int nf = 0; nf < 4; ++nf) {
#pragma unroll
            for (int r = 0; r < 8; ++r) {
                float pv = exp2f((s[nf][r] - mnew[r]) * L2E);
                psum[r] += pv;
                ldsP[wid][r + half * 8][nf * 16 + l16] = (__bf16)pv;
            }
        }
#pragma unroll
        for (int r = 0; r < 8; ++r) {
            float ps = psum[r];
            ps += __shfl_xor(ps, 1);
            ps += __shfl_xor(ps, 2);
            ps += __shfl_xor(ps, 4);
            ps += __shfl_xor(ps, 8);
            rs[r] = rs[r] * corr[r] + ps;
        }
#pragma unroll
        for (int uf = 0; uf < 8; ++uf)
#pragma unroll
            for (int r = 0; r < 8; ++r) o[uf][r] *= corr[r];

        // O += P V
        v16bf ap[2];
#pragma unroll
        for (int c = 0; c < 2; ++c) {
            const __bf16* q = &ldsP[wid][l16][c * 32 + half * 8];
            ap[c] = load_frag2(q, q + 16);
        }
#pragma unroll
        for (int uf = 0; uf < 8; ++uf) {
#pragma unroll
            for (int c = 0; c < 2; ++c) {
                const __bf16* q = &ldsV[p][uf * 16 + l16][c * 32 + half * 8];
                v16bf bv = load_frag2(q, q + 16);
                o[uf] = __builtin_amdgcn_wmma_f32_16x16x32_bf16(
                    false, ap[c], false, bv, (short)0, o[uf], false, false);
            }
        }

        cp_wait();
        __syncthreads();
        p ^= 1;
    }

    // normalize + store
#pragma unroll
    for (int uf = 0; uf < 8; ++uf) {
        int u = uf * 16 + l16;
#pragma unroll
        for (int r = 0; r < 8; ++r) {
            int srow2 = blockIdx.x * 128 + wid * 16 + r + half * 8;
            float v = o[uf][r] / rs[r];
            flat[((long)(b * SEQ + srow2)) * (HEADS * UNITS) + h * UNITS + u] = (__bf16)v;
        }
    }
}

// ---------------------------------------------------------------------------
// launcher
// ---------------------------------------------------------------------------
extern "C" void kernel_launch(void* const* d_in, const int* in_sizes, int n_in,
                              void* d_out, int out_size, void* d_ws, size_t ws_size,
                              hipStream_t stream) {
    const float* x  = (const float*)d_in[0];
    const float* qk = (const float*)d_in[1];
    const float* kk = (const float*)d_in[2];
    const float* vk = (const float*)d_in[3];
    const float* qb = (const float*)d_in[4];
    const float* kb = (const float*)d_in[5];
    const float* vb = (const float*)d_in[6];
    const float* hk = (const float*)d_in[7];
    const float* hb = (const float*)d_in[8];

    char* ws = (char*)d_ws;
    size_t off = 0;
    auto carve = [&](size_t bytes) {
        void* p = ws + off;
        off = (off + bytes + 255) & ~(size_t)255;
        return p;
    };
    __bf16* xb     = (__bf16*)carve((size_t)M_TOTAL * D_IN * 2);
    __bf16* wqkvT  = (__bf16*)carve((size_t)3 * HEADS * UNITS * D_IN * 2);  // [24][128][1024]
    __bf16* wheadT = (__bf16*)carve((size_t)UNITS * HEADS * UNITS * 2);     // [128][1024]
    float*  bqkv   = (float*) carve((size_t)3 * HEADS * UNITS * 4);
    __bf16* qkPl   = (__bf16*)carve((size_t)2 * HEADS * M_TOTAL * UNITS * 2); // q,k planes
    __bf16* vT     = (__bf16*)carve((size_t)HEADS * UNITS * M_TOTAL * 2);     // [h][u][m]
    __bf16* flat   = (__bf16*)carve((size_t)M_TOTAL * HEADS * UNITS * 2);
    (void)ws_size; (void)n_in; (void)in_sizes; (void)out_size;

    // x -> bf16
    {
        int n4 = (M_TOTAL * D_IN) / 4;
        cvt_f32_to_bf16<<<(n4 + 255) / 256, 256, 0, stream>>>(x, xb, n4);
    }
    // weights -> bf16, transposed to [N][K]
    {
        dim3 g(D_IN / 32, UNITS / 32, HEADS);
        transpose_cvt_bf16<<<g, 256, 0, stream>>>(qk, wqkvT + 0L * HEADS * UNITS * D_IN, D_IN, UNITS);
        transpose_cvt_bf16<<<g, 256, 0, stream>>>(kk, wqkvT + 1L * HEADS * UNITS * D_IN, D_IN, UNITS);
        transpose_cvt_bf16<<<g, 256, 0, stream>>>(vk, wqkvT + 2L * HEADS * UNITS * D_IN, D_IN, UNITS);
        dim3 gh((HEADS * UNITS) / 32, UNITS / 32, 1);
        transpose_cvt_bf16<<<gh, 256, 0, stream>>>(hk, wheadT, HEADS * UNITS, UNITS);
    }
    // pack biases [3][H][128]
    (void)hipMemcpyAsync(bqkv + 0 * HEADS * UNITS, qb, HEADS * UNITS * 4, hipMemcpyDeviceToDevice, stream);
    (void)hipMemcpyAsync(bqkv + 1 * HEADS * UNITS, kb, HEADS * UNITS * 4, hipMemcpyDeviceToDevice, stream);
    (void)hipMemcpyAsync(bqkv + 2 * HEADS * UNITS, vb, HEADS * UNITS * 4, hipMemcpyDeviceToDevice, stream);

    // fused QKV projection (V stored transposed)
    gemm_bias_n128<true><<<dim3(M_TOTAL / 128, 3 * HEADS), 256, 0, stream>>>(
        xb, wqkvT, bqkv, qkPl, vT, M_TOTAL, D_IN,
        (long)UNITS * D_IN, (long)UNITS, (long)M_TOTAL * UNITS);

    // flash attention
    attn_kernel<<<dim3(SEQ / 128, BATCH * HEADS), 256, 0, stream>>>(qkPl, vT, flat);

    // head projection -> fp32 d_out
    gemm_bias_n128<false><<<dim3(M_TOTAL / 128, 1), 256, 0, stream>>>(
        flat, wheadT, hb, d_out, nullptr, M_TOTAL, HEADS * UNITS, 0, 0, 0);
}
